// GQAAttention_52364241272913
// MI455X (gfx1250) — matrix operand order repaired and checked by reading, not compiled, not run on hardware
//
#include <hip/hip_runtime.h>
#include <stdint.h>

#define B_   4
#define L_   2048
#define D_   1024
#define H_   16
#define HD_  64
#define NHD_ 1024                // H_*HD_
#define SCALE_   0.125f          // 1/sqrt(64)
#define NEG_INF_ (-1.0e9f)

typedef __bf16 v16bf __attribute__((ext_vector_type(16)));
typedef float  v8f   __attribute__((ext_vector_type(8)));

struct alignas(16) U4 { unsigned int a, b, c, d; };
union Frag { v16bf v; U4 u[2]; };

static __device__ __forceinline__ v8f wmma_bf16(const Frag& a, const Frag& b, v8f c) {
    return __builtin_amdgcn_wmma_f32_16x16x32_bf16(false, a.v, false, b.v,
                                                   (short)0, c, false, false);
}

// ---------------------------------------------------------------------------
// f32 -> bf16 elementwise convert
__global__ void cvt_bf16_k(const float* __restrict__ x, __bf16* __restrict__ y, int n) {
    int i = blockIdx.x * blockDim.x + threadIdx.x;
    if (i < n) y[i] = (__bf16)x[i];
}

// f32 W[K][N] -> bf16 Wt[N][K]  (so GEMM B-fragments are K-contiguous)
__global__ void cvt_tr_bf16_k(const float* __restrict__ W, __bf16* __restrict__ Wt,
                              int K, int N) {
    int i = blockIdx.x * blockDim.x + threadIdx.x;   // output index n*K + k
    if (i < K * N) {
        int k = i % K;
        int n = i / K;
        Wt[i] = (__bf16)W[(size_t)k * N + n];
    }
}

// ---------------------------------------------------------------------------
// C[M][N] f32 = A[M][K]bf16 @ Bt[N][K]bf16^T + bias[N]
// one wave -> 16 rows x 64 cols (4 WMMA column tiles)
__global__ void gemm_bf16_k(const __bf16* __restrict__ A,
                            const __bf16* __restrict__ Bt,
                            const float*  __restrict__ bias,
                            float* __restrict__ C,
                            int M, int N, int K) {
    const int lane = threadIdx.x & 31;
    const int warp = threadIdx.x >> 5;
    const int wid  = blockIdx.x * 4 + warp;
    const int rowTiles = M >> 4;
    const int colGrps  = N >> 6;
    if (wid >= rowTiles * colGrps) return;
    const int tileM  = wid % rowTiles;
    const int colGrp = wid / rowTiles;
    const int n15  = lane & 15;
    const int half = lane >> 4;

    const __bf16* Arow = A + (size_t)(tileM * 16 + n15) * K;
    v8f acc[4] = {};

    for (int kb = 0; kb < K; kb += 32) {
        __builtin_prefetch(Arow + kb + 128, 0, 1);
        Frag a;
        a.u[0] = *(const U4*)(Arow + kb + 8 * half);        // K = 8h+0..7
        a.u[1] = *(const U4*)(Arow + kb + 16 + 8 * half);   // K = 16+8h+0..7
#pragma unroll
        for (int c = 0; c < 4; c++) {
            const __bf16* Brow = Bt + (size_t)(colGrp * 64 + c * 16 + n15) * K;
            Frag b;
            b.u[0] = *(const U4*)(Brow + kb + 8 * half);        // K = 8h+0..7
            b.u[1] = *(const U4*)(Brow + kb + 16 + 8 * half);   // K = 16+8h+0..7
            acc[c] = wmma_bf16(a, b, acc[c]);
        }
    }
#pragma unroll
    for (int c = 0; c < 4; c++) {
        const int col = colGrp * 64 + c * 16 + n15;
        const float bs = bias[col];
#pragma unroll
        for (int v = 0; v < 8; v++) {
            const int row = tileM * 16 + v + 8 * half;
            C[(size_t)row * N + col] = acc[c][v] + bs;
        }
    }
}

// ---------------------------------------------------------------------------
// RoPE + head-major repack.
// mode 0: Q -> dst[(b*H+h)*L + l][d], rope, *SCALE
// mode 1: K -> same layout, rope
// mode 2: V -> dstT[((b*H+h)*64 + d)*L + l], no rope
__global__ void rope_pack_k(const float* __restrict__ src, int srcStride, int srcColOff,
                            const float* __restrict__ rope,
                            __bf16* __restrict__ dst, int mode, float scale) {
    int t = blockIdx.x * blockDim.x + threadIdx.x;
    if (t >= B_ * H_ * L_ * 32) return;
    const int p = t & 31;            // rotary pair 0..31
    const int l = (t >> 5) & (L_ - 1);
    const int h = (t >> 16) & (H_ - 1);
    const int b = t >> 20;

    const size_t si = ((size_t)(b * L_ + l)) * srcStride + srcColOff + h * 64 + 2 * p;
    const float xe = src[si], xo = src[si + 1];
    float re, ro;
    if (mode < 2) {
        const float sn = rope[l * HD_ + 2 * p];
        const float cs = rope[l * HD_ + 2 * p + 1];
        re = (xe * cs - xo * sn) * scale;
        ro = (xe * sn + xo * cs) * scale;
    } else { re = xe; ro = xo; }

    if (mode < 2) {
        const size_t di = (((size_t)(b * H_ + h)) * L_ + l) * HD_ + 2 * p;
        dst[di]     = (__bf16)re;
        dst[di + 1] = (__bf16)ro;
    } else {
        const size_t rb = ((size_t)(b * H_ + h)) * HD_;
        dst[(rb + 2 * p) * L_ + l]     = (__bf16)re;
        dst[(rb + 2 * p + 1) * L_ + l] = (__bf16)ro;
    }
}

// ---------------------------------------------------------------------------
// Flash attention, causal. One wave per (b,h, 16-query tile); 32 keys/step.
// Qh,Kh: [B*H][L][64] bf16 (Q pre-scaled).  Vt: [B*H][64][L] bf16.
// Out: [B][L][H*64] bf16.
__global__ void flash_attn_k(const __bf16* __restrict__ Qh,
                             const __bf16* __restrict__ Kh,
                             const __bf16* __restrict__ Vt,
                             __bf16* __restrict__ Oattn) {
    __shared__ __bf16 Pl[4][16 * 32];
    const int lane = threadIdx.x & 31;
    const int warp = threadIdx.x >> 5;
    const int wid  = blockIdx.x * 4 + warp;
    const int qTiles = L_ / 16;
    const int qt = wid % qTiles;
    const int bh = wid / qTiles;
    if (bh >= B_ * H_) return;
    const int qBase = qt * 16;
    const int n15  = lane & 15;
    const int half = lane >> 4;

    const __bf16* Qp = Qh + (size_t)bh * L_ * HD_;
    const __bf16* Kp = Kh + (size_t)bh * L_ * HD_;
    const __bf16* Vp = Vt + (size_t)bh * HD_ * L_;

    // Q A-fragments for d=0..31 and d=32..63
    Frag aQ[2];
    const __bf16* qrow = Qp + (size_t)(qBase + n15) * HD_;
    aQ[0].u[0] = *(const U4*)(qrow + 8 * half);
    aQ[0].u[1] = *(const U4*)(qrow + 16 + 8 * half);
    aQ[1].u[0] = *(const U4*)(qrow + 32 + 8 * half);
    aQ[1].u[1] = *(const U4*)(qrow + 48 + 8 * half);

    float mSt[8], lSt[8];
    v8f accO[4] = {};
#pragma unroll
    for (int v = 0; v < 8; v++) { mSt[v] = -3.0e38f; lSt[v] = 0.0f; }
    __bf16* Pw = &Pl[warp][0];

    const int nCh = (qBase + 47) >> 5;               // ceil((qBase+16)/32)
    for (int c = 0; c < nCh; c++) {
        const int kBase = c * 32;
        // S = Q @ K^T for keys kBase..kBase+31 (two 16-key column tiles)
        v8f s0 = {}, s1 = {};
        {
            const __bf16* kr0 = Kp + (size_t)(kBase + n15) * HD_;
            const __bf16* kr1 = Kp + (size_t)(kBase + 16 + n15) * HD_;
            Frag b;
            b.u[0] = *(const U4*)(kr0 + 8 * half);
            b.u[1] = *(const U4*)(kr0 + 16 + 8 * half);
            s0 = wmma_bf16(aQ[0], b, s0);
            b.u[0] = *(const U4*)(kr0 + 32 + 8 * half);
            b.u[1] = *(const U4*)(kr0 + 48 + 8 * half);
            s0 = wmma_bf16(aQ[1], b, s0);
            b.u[0] = *(const U4*)(kr1 + 8 * half);
            b.u[1] = *(const U4*)(kr1 + 16 + 8 * half);
            s1 = wmma_bf16(aQ[0], b, s1);
            b.u[0] = *(const U4*)(kr1 + 32 + 8 * half);
            b.u[1] = *(const U4*)(kr1 + 48 + 8 * half);
            s1 = wmma_bf16(aQ[1], b, s1);
        }
        // online softmax (row r = v + 8*half lives in one 16-lane half)
#pragma unroll
        for (int v = 0; v < 8; v++) {
            const int qq = qBase + v + 8 * half;
            float s0v = s0[v]; if (kBase + n15 > qq)      s0v = NEG_INF_;
            float s1v = s1[v]; if (kBase + 16 + n15 > qq) s1v = NEG_INF_;
            float rmx = fmaxf(s0v, s1v);
#pragma unroll
            for (int off = 1; off < 16; off <<= 1)
                rmx = fmaxf(rmx, __shfl_xor(rmx, off, 32));
            const float mN = fmaxf(mSt[v], rmx);
            const float p0 = __expf(s0v - mN);
            const float p1 = __expf(s1v - mN);
            float rs = p0 + p1;
#pragma unroll
            for (int off = 1; off < 16; off <<= 1)
                rs += __shfl_xor(rs, off, 32);
            const float alpha = __expf(mSt[v] - mN);
            mSt[v] = mN;
            lSt[v] = lSt[v] * alpha + rs;
            accO[0][v] *= alpha; accO[1][v] *= alpha;
            accO[2][v] *= alpha; accO[3][v] *= alpha;
            const int r = v + 8 * half;
            Pw[r * 32 + n15]      = (__bf16)p0;
            Pw[r * 32 + 16 + n15] = (__bf16)p1;
        }
        // C-layout -> A-layout via per-wave LDS tile (same-wave DS ordering)
        asm volatile("s_wait_dscnt 0" ::: "memory");
        Frag aP;
        aP.u[0] = *(const U4*)(Pw + n15 * 32 + 8 * half);
        aP.u[1] = *(const U4*)(Pw + n15 * 32 + 16 + 8 * half);
        // O += P @ V   (V^T rows are key-contiguous -> B128 loads)
#pragma unroll
        for (int dc = 0; dc < 4; dc++) {
            const __bf16* vr = Vp + (size_t)(dc * 16 + n15) * L_ + kBase + 8 * half;
            Frag bV;
            bV.u[0] = *(const U4*)(vr);
            bV.u[1] = *(const U4*)(vr + 16);
            accO[dc] = wmma_bf16(aP, bV, accO[dc]);
        }
    }
    // epilogue: normalize and store [B][L][H*64] bf16
    const int b = bh / H_, h = bh % H_;
#pragma unroll
    for (int v = 0; v < 8; v++) {
        const float inv = 1.0f / lSt[v];
        const int l = qBase + v + 8 * half;
        const size_t o = ((size_t)(b * L_ + l)) * NHD_ + h * HD_;
#pragma unroll
        for (int dc = 0; dc < 4; dc++)
            Oattn[o + dc * 16 + n15] = (__bf16)(accO[dc][v] * inv);
    }
}

// ---------------------------------------------------------------------------
extern "C" void kernel_launch(void* const* d_in, const int* in_sizes, int n_in,
                              void* d_out, int out_size, void* d_ws, size_t ws_size,
                              hipStream_t stream) {
    const float* q_in = (const float*)d_in[0];
    const float* k_in = (const float*)d_in[1];
    // d_in[2] (v_in) unused by reference; d_in[3] (mask) is causal tril -> applied analytically
    const float* rope = (const float*)d_in[4];
    const float* Wq   = (const float*)d_in[5];
    const float* bq   = (const float*)d_in[6];
    const float* Wkv  = (const float*)d_in[7];
    const float* bkv  = (const float*)d_in[8];
    const float* Wp   = (const float*)d_in[9];
    const float* bp   = (const float*)d_in[10];

    const size_t BLD = (size_t)B_ * L_ * D_;   // 8,388,608 (== B*H*L*HD)
    uintptr_t cur = (uintptr_t)d_ws;
    auto take = [&](size_t bytes) -> uintptr_t {
        uintptr_t p = cur; cur += (bytes + 255) & ~(size_t)255; return p;
    };
    __bf16* qb   = (__bf16*)take(BLD * 2);
    __bf16* kb   = (__bf16*)take(BLD * 2);
    __bf16* WqT  = (__bf16*)take((size_t)D_ * NHD_ * 2);
    __bf16* WkvT = (__bf16*)take((size_t)D_ * 2 * NHD_ * 2);
    __bf16* WpT  = (__bf16*)take((size_t)NHD_ * D_ * 2);
    float*  Qf   = (float*) take(BLD * 4);
    float*  KVf  = (float*) take(BLD * 2 * 4);
    __bf16* Qh   = (__bf16*)take(BLD * 2);
    __bf16* Kh   = (__bf16*)take(BLD * 2);
    __bf16* Vt   = (__bf16*)take(BLD * 2);
    __bf16* Ab   = (__bf16*)((uintptr_t)Qf);   // reuse Qf region (dead after rope pack)

    const int T = 256;
    // 1) convert activations + transpose-convert weights to bf16
    cvt_bf16_k<<<(int)((BLD + T - 1) / T), T, 0, stream>>>(q_in, qb, (int)BLD);
    cvt_bf16_k<<<(int)((BLD + T - 1) / T), T, 0, stream>>>(k_in, kb, (int)BLD);
    cvt_tr_bf16_k<<<(D_ * NHD_ + T - 1) / T, T, 0, stream>>>(Wq,  WqT,  D_, NHD_);
    cvt_tr_bf16_k<<<(D_ * 2 * NHD_ + T - 1) / T, T, 0, stream>>>(Wkv, WkvT, D_, 2 * NHD_);
    cvt_tr_bf16_k<<<(NHD_ * D_ + T - 1) / T, T, 0, stream>>>(Wp,  WpT,  NHD_, D_);

    // 2) projections (WMMA GEMMs)
    const int M = B_ * L_;                       // 8192
    gemm_bf16_k<<<(M / 16) * (NHD_ / 64) / 4, 128, 0, stream>>>(qb, WqT, bq, Qf, M, NHD_, D_);
    gemm_bf16_k<<<(M / 16) * (2 * NHD_ / 64) / 4, 128, 0, stream>>>(kb, WkvT, bkv, KVf, M, 2 * NHD_, D_);

    // 3) RoPE + head-major repack (Q pre-scaled by 1/sqrt(HD); V transposed)
    const int RT = B_ * H_ * L_ * 32;
    rope_pack_k<<<RT / T, T, 0, stream>>>(Qf, NHD_, 0, rope, Qh, 0, SCALE_);
    rope_pack_k<<<RT / T, T, 0, stream>>>(KVf, 2 * NHD_, 0, rope, Kh, 1, 1.0f);
    rope_pack_k<<<RT / T, T, 0, stream>>>(KVf, 2 * NHD_, NHD_, rope, Vt, 2, 1.0f);

    // 4) causal flash attention (WMMA QK^T + PV, online softmax)
    flash_attn_k<<<B_ * H_ * (L_ / 16) / 4, 128, 0, stream>>>(Qh, Kh, Vt, Ab);

    // 5) output projection -> f32 d_out
    gemm_bf16_k<<<(M / 16) * (D_ / 64) / 4, 128, 0, stream>>>(Ab, WpT, bp, (float*)d_out,
                                                             M, D_, NHD_);
}
